// MyInterleavedModule_14525579395117
// MI455X (gfx1250) — compile-verified, matrix operand order checked by reading, not run on hardware
//
#include <hip/hip_runtime.h>
#include <stdint.h>

// ---------------------------------------------------------------------------
// C[B,N] = X[B,K] @ W[N,K]^T ; B=524288, K=256, N=256, fp32 in/out.
// bf16 WMMA (V_WMMA_F32_16X16X32_BF16) with fp32 accumulate, LDS-staged,
// software-pipelined global loads, NT hints on the X/C streams,
// all 8 B fragments preloaded per chunk for staggered s_wait_dscnt.
// MI455X / gfx1250, wave32.
// ---------------------------------------------------------------------------

#define BATCH   524288
#define KDIM    256
#define NDIM    256
#define BM      128          // batch rows per workgroup
#define KC      32           // K chunk (one WMMA K step)
#define THREADS 512          // 16 wave32 waves
#define NKCHUNK (KDIM / KC)  // 8

// LDS row padding: 32 data bf16 + 8 pad = 40 elems = 80 B = 20 dwords.
// Bank math (64 banks x 4B): row r -> bank (20*r) % 64, distinct for r=0..15,
// and the +16B K-half lands on the next 4-bank group => conflict-free b128.
#define LROW_DW 20

typedef __attribute__((ext_vector_type(16))) __bf16 v16bf;
typedef __attribute__((ext_vector_type(4)))  __bf16 v4bf;
typedef __attribute__((ext_vector_type(8)))  float  v8f;
typedef __attribute__((ext_vector_type(4)))  float  v4f;   // true clang vector
typedef __attribute__((ext_vector_type(4)))  uint32_t v4u;

__device__ __forceinline__ void stage_quad(uint32_t* dst, v4f f) {
    // Native f32->bf16 converts (RNE) -> v_cvt_pk_bf16_f32, one ds_store_b64.
    v4bf v;
    v.x = (__bf16)f.x;
    v.y = (__bf16)f.y;
    v.z = (__bf16)f.z;
    v.w = (__bf16)f.w;
    *(v4bf*)dst = v;
}

__device__ __forceinline__ v16bf lds_frag(const uint32_t* p) {
    // Two 16B chunks: K 0..7 (this half) and K 16..23 (this half) per ISA
    // 16-bit A-matrix layout; p is 16B-aligned into LDS.
    union { v4u u[2]; v16bf v; } t;
    t.u[0] = *(const v4u*)(p);
    t.u[1] = *(const v4u*)(p + 8);
    return t.v;
}

__global__ void __launch_bounds__(THREADS, 1)
gemm_bf16_wmma(const float* __restrict__ x,
               const float* __restrict__ w,
               float* __restrict__ out) {
    __shared__ uint32_t sX[BM   * LROW_DW];  // 128 rows * 80B = 10 KB
    __shared__ uint32_t sW[NDIM * LROW_DW];  // 256 rows * 80B = 20 KB

    const int tid     = threadIdx.x;
    const int rowBase = blockIdx.x * BM;

    // ---------------- loader mapping (all threads) -------------------------
    // One "quad" = 4 consecutive floats (16B) of one row of the K-chunk.
    // X: 128 rows * 8 quads = 1024 quads -> 2/thread (NT stream loads).
    // W: 256 rows * 8 quads = 2048 quads -> 4/thread (RT, stays in L2).
    const float* gX[2];  uint32_t* lX[2];
    const float* gW[4];  uint32_t* lW[4];
#pragma unroll
    for (int i = 0; i < 2; ++i) {
        const int q = tid + i * THREADS;
        const int r = q >> 3, c = q & 7;
        gX[i] = x + (size_t)(rowBase + r) * KDIM + c * 4;
        lX[i] = &sX[r * LROW_DW + c * 2];
    }
#pragma unroll
    for (int i = 0; i < 4; ++i) {
        const int q = tid + i * THREADS;
        const int r = q >> 3, c = q & 7;
        gW[i] = w + (size_t)r * KDIM + c * 4;
        lW[i] = &sW[r * LROW_DW + c * 2];
    }

    // ---------------- compute mapping (per wave32) -------------------------
    const int wave  = tid >> 5;          // 0..15
    const int lane  = tid & 31;
    const int row16 = lane & 15;         // M (or N) within a 16-tile
    const int kh    = lane >> 4;         // K-half select
    const int mStrip = (wave >> 1) * 16;         // 8 row strips of 16
    const int nHalf  = (wave & 1) * 128;         // 2 column halves of 128

    const uint32_t* aPtr  = &sX[(mStrip + row16) * LROW_DW + kh * 4];
    const uint32_t* bBase = &sW[(nHalf + row16) * LROW_DW + kh * 4];

    v8f acc[8];
#pragma unroll
    for (int t = 0; t < 8; ++t) acc[t] = (v8f){0.f,0.f,0.f,0.f,0.f,0.f,0.f,0.f};

    // ---------------- software-pipelined K loop ----------------------------
    v4f preX[2], preW[4];
#pragma unroll
    for (int i = 0; i < 2; ++i)
        preX[i] = __builtin_nontemporal_load((const v4f*)gX[i]);
#pragma unroll
    for (int i = 0; i < 4; ++i)
        preW[i] = *(const v4f*)gW[i];

#pragma unroll
    for (int kc = 0; kc < NKCHUNK; ++kc) {
        if (kc) __syncthreads();               // prior compute done with LDS

        // convert + stage this chunk into LDS (native bf16 converts)
#pragma unroll
        for (int i = 0; i < 2; ++i) stage_quad(lX[i], preX[i]);
#pragma unroll
        for (int i = 0; i < 4; ++i) stage_quad(lW[i], preW[i]);
        __syncthreads();

        // issue next chunk's global loads; they fly during the WMMAs below
        if (kc + 1 < NKCHUNK) {
            const int koff = (kc + 1) * KC;    // +32 floats along the row
#pragma unroll
            for (int i = 0; i < 2; ++i)
                preX[i] = __builtin_nontemporal_load((const v4f*)(gX[i] + koff));
#pragma unroll
            for (int i = 0; i < 4; ++i)
                preW[i] = *(const v4f*)(gW[i] + koff);
        }

        // 8 WMMA tiles: D = A(16x32 bf16) x B(32x16 bf16) + C(f32).
        // Preload ALL 8 B fragments (16x ds_load_b128) before the first WMMA
        // so the WMMAs retire against staggered s_wait_dscnt (0xe..0x0)
        // instead of serializing load->wait0->wmma.
        const v16bf a = lds_frag(aPtr);
        v16bf b[8];
#pragma unroll
        for (int j = 0; j < 8; ++j)
            b[j] = lds_frag(bBase + j * 16 * LROW_DW);
#pragma unroll
        for (int nt = 0; nt < 8; ++nt) {
            acc[nt] = __builtin_amdgcn_wmma_f32_16x16x32_bf16(
                /*neg_a=*/false, a, /*neg_b=*/false, b[nt],
                /*c_mod=*/(short)0, acc[nt],
                /*reuse_a=*/false, /*reuse_b=*/false);
        }
    }

    // ---------------- store C (NT: write-once stream) -----------------------
    // f32 16x16 C/D layout: VGPR r -> (M=r, N=lane) lanes 0-15,
    //                                 (M=8+r, N=lane-16) lanes 16-31.
    const int mBase = rowBase + mStrip + kh * 8;
#pragma unroll
    for (int nt = 0; nt < 8; ++nt) {
        const int col = nHalf + nt * 16 + row16;
#pragma unroll
        for (int r = 0; r < 8; ++r) {
            __builtin_nontemporal_store(acc[nt][r],
                                        &out[(size_t)(mBase + r) * NDIM + col]);
        }
    }
}

extern "C" void kernel_launch(void* const* d_in, const int* in_sizes, int n_in,
                              void* d_out, int out_size, void* d_ws, size_t ws_size,
                              hipStream_t stream) {
    (void)in_sizes; (void)n_in; (void)out_size; (void)d_ws; (void)ws_size;
    const float* x = (const float*)d_in[0];   // [BATCH, KDIM] fp32
    const float* w = (const float*)d_in[1];   // [NDIM,  KDIM] fp32
    float*     out = (float*)d_out;           // [BATCH, NDIM] fp32

    dim3 grid(BATCH / BM);
    dim3 block(THREADS);
    gemm_bf16_wmma<<<grid, block, 0, stream>>>(x, w, out);
}
